// QCompass_45758581571589
// MI455X (gfx1250) — matrix operand order repaired and checked by reading, not compiled, not run on hardware
//
#include <hip/hip_runtime.h>
#include <cmath>

typedef __bf16 bf16_t;
typedef __attribute__((ext_vector_type(16))) __bf16 v16bf;
typedef __attribute__((ext_vector_type(8)))  __bf16 v8bf;
typedef __attribute__((ext_vector_type(8)))  float  v8f;
typedef __attribute__((ext_vector_type(4)))  float  f32x4;

// ---------------------------------------------------------------------------
// WMMA helper: D = A(16x32 bf16) * B(32x16 bf16) + C(16x16 f32)
// ---------------------------------------------------------------------------
__device__ __forceinline__ v8f wmma_bf16(v16bf a, v16bf b, v8f c) {
  return __builtin_amdgcn_wmma_f32_16x16x32_bf16(
      /*neg_a=*/false, a, /*neg_b=*/false, b,
      /*c_mod=*/(short)0, c, /*reuse_a=*/false, /*reuse_b=*/false);
}

// A fragment (16x32 bf16) from row-major [rows, ld] bf16.
// Lanes 0-15 (M=lane): K = k0+{0..7, 16..23}; lanes 16-31: K = k0+{8..15, 24..31}
__device__ __forceinline__ v16bf load_a_bf16(const bf16_t* __restrict__ src, long ld,
                                             int row0, int k0) {
  int lane = threadIdx.x & 31;
  int half = lane >> 4;
  long r = row0 + (lane & 15);
  const bf16_t* p = src + r * ld + k0 + half * 8;
  v8bf lo = *(const v8bf*)(p);
  v8bf hi = *(const v8bf*)(p + 16);
  return __builtin_shufflevector(lo, hi, 0,1,2,3,4,5,6,7,8,9,10,11,12,13,14,15);
}

// A fragment with on-the-fly fp32 -> bf16 conversion (row-major fp32 source).
__device__ __forceinline__ v16bf load_a_f32(const float* __restrict__ src, long ld,
                                            int row0, int k0) {
  int lane = threadIdx.x & 31;
  int half = lane >> 4;
  long r = row0 + (lane & 15);
  const float* p = src + r * ld + k0 + half * 8;
  f32x4 a0 = *(const f32x4*)(p);
  f32x4 a1 = *(const f32x4*)(p + 4);
  f32x4 a2 = *(const f32x4*)(p + 16);
  f32x4 a3 = *(const f32x4*)(p + 20);
  v16bf f;
#pragma unroll
  for (int i = 0; i < 4; ++i) {
    f[i]      = (bf16_t)a0[i];
    f[4 + i]  = (bf16_t)a1[i];
    f[8 + i]  = (bf16_t)a2[i];
    f[12 + i] = (bf16_t)a3[i];
  }
  return f;
}

// B fragment (32x16 bf16) from B^T stored row-major [N, ld] bf16.
// Lanes 0-15 (N=lane): K = k0+{0..7, 16..23}; lanes 16-31: K = k0+{8..15, 24..31}
__device__ __forceinline__ v16bf load_b_bf16(const bf16_t* __restrict__ srcT, long ld,
                                             int n0, int k0) {
  int lane = threadIdx.x & 31;
  int half = lane >> 4;
  long n = n0 + (lane & 15);
  const bf16_t* p = srcT + n * ld + k0 + half * 8;
  v8bf lo = *(const v8bf*)(p);
  v8bf hi = *(const v8bf*)(p + 16);
  return __builtin_shufflevector(lo, hi, 0,1,2,3,4,5,6,7,8,9,10,11,12,13,14,15);
}

__device__ __forceinline__ void store_c_bf16(bf16_t* __restrict__ dst, long ld,
                                             int m0, int n0, v8f c, float scale) {
  int lane = threadIdx.x & 31;
  int col = lane & 15;
  int ro = (lane >> 4) * 8;
#pragma unroll
  for (int v = 0; v < 8; ++v)
    dst[(long)(m0 + v + ro) * ld + n0 + col] = (bf16_t)(c[v] * scale);
}

__device__ __forceinline__ void store_c_f32_bias(float* __restrict__ dst, long ld,
                                                 int m0, int n0, v8f c,
                                                 const float* __restrict__ bo) {
  int lane = threadIdx.x & 31;
  int col = lane & 15;
  int ro = (lane >> 4) * 8;
  float bias = bo[n0 + col];
#pragma unroll
  for (int v = 0; v < 8; ++v)
    dst[(long)(m0 + v + ro) * ld + n0 + col] = c[v] + bias;
}

// ---------------------------------------------------------------------------
// Prep kernels
// ---------------------------------------------------------------------------
__global__ void __launch_bounds__(256)
k_prep_wt(const float* __restrict__ Ws, const float* __restrict__ Wa,
          bf16_t* __restrict__ WsT, bf16_t* __restrict__ WaT, int H, int R) {
  const float* src = blockIdx.y ? Wa : Ws;
  bf16_t* dst = blockIdx.y ? WaT : WsT;
  int tid = blockIdx.x * blockDim.x + threadIdx.x;
  if (tid >= H * R) return;
  int h = tid / R, r = tid % R;
  dst[(long)r * H + h] = (bf16_t)src[tid];
}

__global__ void __launch_bounds__(256)
k_prep_wo(const float* __restrict__ Wo, bf16_t* __restrict__ Wob, int n) {
  int tid = blockIdx.x * blockDim.x + threadIdx.x;
  if (tid < n) Wob[tid] = (bf16_t)Wo[tid];
}

// x[b] [L, H] fp32 -> xT [H, L] bf16 (tiled LDS transpose)
__global__ void __launch_bounds__(256)
k_transpose(const float* __restrict__ x, bf16_t* __restrict__ xt, int H, int L) {
  __shared__ bf16_t t[32][33];
  int l0 = blockIdx.x * 32, h0 = blockIdx.y * 32;
  for (int r = threadIdx.y; r < 32; r += 8)
    t[r][threadIdx.x] = (bf16_t)x[(long)(l0 + r) * H + h0 + threadIdx.x];
  __syncthreads();
  for (int r = threadIdx.y; r < 32; r += 8)
    xt[(long)(h0 + r) * L + l0 + threadIdx.x] = t[threadIdx.x][r];
}

// ---------------------------------------------------------------------------
// Projections: state = (x @ Ws)/16, action = x @ Wa.  M=B*L, K=H, N=R.
// One wave = 64x64 output tile; A (fp32) loaded+converted just-in-time so
// only one fp32 transient set is live at a time.
// ---------------------------------------------------------------------------
__global__ void __launch_bounds__(32)
k_proj(const float* __restrict__ x, const bf16_t* __restrict__ WsT,
       const bf16_t* __restrict__ WaT, bf16_t* __restrict__ state,
       bf16_t* __restrict__ action, int K, int N, float inv_scale) {
  const bf16_t* Wt = blockIdx.z ? WaT : WsT;
  bf16_t* dst = blockIdx.z ? action : state;
  float scale = blockIdx.z ? 1.0f : inv_scale;
  int m0 = blockIdx.x * 64, n0 = blockIdx.y * 64;
  v8f c[4][4] = {};
  for (int k0 = 0; k0 < K; k0 += 32) {
    v16bf b[4];
#pragma unroll
    for (int ni = 0; ni < 4; ++ni) b[ni] = load_b_bf16(Wt, K, n0 + 16 * ni, k0);
#pragma unroll
    for (int mi = 0; mi < 4; ++mi) {
      v16bf a = load_a_f32(x, K, m0 + 16 * mi, k0);
#pragma unroll
      for (int ni = 0; ni < 4; ++ni) c[mi][ni] = wmma_bf16(a, b[ni], c[mi][ni]);
    }
  }
#pragma unroll
  for (int mi = 0; mi < 4; ++mi)
#pragma unroll
    for (int ni = 0; ni < 4; ++ni)
      store_c_bf16(dst, N, m0 + 16 * mi, n0 + 16 * ni, c[mi][ni], scale);
}

// ---------------------------------------------------------------------------
// Causal scores + softmax -> P (bf16), one batch.
// 4-wave block per 16-row query tile; wave w handles j-tiles with
// (j0/16)%4 == w. Pass 1: partial online (m,l); merge via LDS (exact);
// pass 2: each wave writes its quarter of the row, zero-padded to the
// 64-aligned boundary needed by the 64-row PV GEMM blocks.
// ---------------------------------------------------------------------------
__global__ void __launch_bounds__(128)
k_scores(const bf16_t* __restrict__ state, const bf16_t* __restrict__ action,
         bf16_t* __restrict__ P, int L, int R) {
  __shared__ float sm[4][16], sl[4][16];
  int w = threadIdx.x >> 5;
  int lane = threadIdx.x & 31;
  int i0 = blockIdx.x * 16;
  int col = lane & 15;
  int ro = (lane >> 4) * 8;

  v16bf a[8];
#pragma unroll
  for (int kk = 0; kk < 8; ++kk) a[kk] = load_a_bf16(state, R, i0, kk * 32);

  float m[8], l[8];
#pragma unroll
  for (int v = 0; v < 8; ++v) { m[v] = -__builtin_inff(); l[v] = 0.0f; }

  // Pass 1: partial row max / sum over this wave's j-tiles
  for (int j0 = w * 16; j0 <= i0; j0 += 64) {
    v8f c = {};
#pragma unroll
    for (int kk = 0; kk < 8; ++kk)
      c = wmma_bf16(a[kk], load_b_bf16(action, R, j0, kk * 32), c);
#pragma unroll
    for (int v = 0; v < 8; ++v) {
      float s = (j0 + col <= i0 + v + ro) ? c[v] : -__builtin_inff();
      float t = s;
#pragma unroll
      for (int off = 1; off < 16; off <<= 1) t = fmaxf(t, __shfl_xor(t, off, 32));
      float mn = fmaxf(m[v], t);
      float e = __expf(s - mn);
#pragma unroll
      for (int off = 1; off < 16; off <<= 1) e += __shfl_xor(e, off, 32);
      l[v] = l[v] * __expf(m[v] - mn) + e;
      m[v] = mn;
    }
  }

  // Exchange partial stats through LDS
  if (col == 0) {
#pragma unroll
    for (int v = 0; v < 8; ++v) { sm[w][ro + v] = m[v]; sl[w][ro + v] = l[v]; }
  }
  __syncthreads();

  float M[8], rl[8];
#pragma unroll
  for (int v = 0; v < 8; ++v) {
    float mm = sm[0][ro + v];
#pragma unroll
    for (int ww = 1; ww < 4; ++ww) mm = fmaxf(mm, sm[ww][ro + v]);
    float s = 0.0f;
#pragma unroll
    for (int ww = 0; ww < 4; ++ww) s += sl[ww][ro + v] * __expf(sm[ww][ro + v] - mm);
    M[v] = mm;
    rl[v] = 1.0f / s;
  }

  // Pass 2: normalized probabilities; zeros past the diagonal, padded so the
  // covering 64-row PV block (K extent = floor64(i0)+64) only reads written P.
  int jend = (i0 / 64) * 64 + 64;
  for (int j0 = w * 16; j0 < jend; j0 += 64) {
    if (j0 <= i0) {
      v8f c = {};
#pragma unroll
      for (int kk = 0; kk < 8; ++kk)
        c = wmma_bf16(a[kk], load_b_bf16(action, R, j0, kk * 32), c);
#pragma unroll
      for (int v = 0; v < 8; ++v) {
        float s = (j0 + col <= i0 + v + ro) ? c[v] : -__builtin_inff();
        float p = __expf(s - M[v]) * rl[v];
        P[(long)(i0 + v + ro) * L + j0 + col] = (bf16_t)p;
      }
    } else {
#pragma unroll
      for (int v = 0; v < 8; ++v)
        P[(long)(i0 + v + ro) * L + j0 + col] = (bf16_t)0.0f;
    }
  }
}

// ---------------------------------------------------------------------------
// out = P @ x  (bf16, x via transposed xT), one batch; causal K bound.
// One wave = 64x64 tile; prefetch streams 128 cols ahead.
// ---------------------------------------------------------------------------
__global__ void __launch_bounds__(32)
k_pv(const bf16_t* __restrict__ P, const bf16_t* __restrict__ xT,
     bf16_t* __restrict__ outb, int L, int H) {
  int i0 = blockIdx.x * 64, h0 = blockIdx.y * 64;
  int lane = threadIdx.x & 31;
  int nk = i0 / 32 + 2;  // K runs over j < i0+64 (P zero-padded to that bound)
  int kend = nk * 32;
  v8f c[4][4] = {};
  for (int t = 0; t < nk; ++t) {
    int k0 = t * 32;
    int kpf = k0 + 128;
    if (((t & 1) == 0) && kpf < kend) {
      __builtin_prefetch(P + (long)(i0 + lane) * L + kpf, 0, 1);
      __builtin_prefetch(P + (long)(i0 + 32 + lane) * L + kpf, 0, 1);
      __builtin_prefetch(xT + (long)(h0 + lane) * L + kpf, 0, 1);
      __builtin_prefetch(xT + (long)(h0 + 32 + lane) * L + kpf, 0, 1);
    }
    v16bf b[4];
#pragma unroll
    for (int ni = 0; ni < 4; ++ni) b[ni] = load_b_bf16(xT, L, h0 + 16 * ni, k0);
#pragma unroll
    for (int mi = 0; mi < 4; ++mi) {
      v16bf a = load_a_bf16(P, L, i0 + 16 * mi, k0);
#pragma unroll
      for (int ni = 0; ni < 4; ++ni) c[mi][ni] = wmma_bf16(a, b[ni], c[mi][ni]);
    }
  }
#pragma unroll
  for (int mi = 0; mi < 4; ++mi)
#pragma unroll
    for (int ni = 0; ni < 4; ++ni)
      store_c_bf16(outb, H, i0 + 16 * mi, h0 + 16 * ni, c[mi][ni], 1.0f);
}

// ---------------------------------------------------------------------------
// y = out @ Wo_eff^T + bo   (64x64 per wave, fp32 output)
// ---------------------------------------------------------------------------
__global__ void __launch_bounds__(32)
k_out(const bf16_t* __restrict__ outb, const bf16_t* __restrict__ Wob,
      const float* __restrict__ bo, float* __restrict__ y, int H) {
  int m0 = blockIdx.x * 64, n0 = blockIdx.y * 64;
  int lane = threadIdx.x & 31;
  v8f c[4][4] = {};
  for (int k0 = 0; k0 < H; k0 += 32) {
    int kpf = k0 + 128;
    if ((((k0 >> 5) & 1) == 0) && kpf < H) {
      __builtin_prefetch(outb + (long)(m0 + lane) * H + kpf, 0, 1);
      __builtin_prefetch(outb + (long)(m0 + 32 + lane) * H + kpf, 0, 1);
      __builtin_prefetch(Wob + (long)(n0 + lane) * H + kpf, 0, 1);
      __builtin_prefetch(Wob + (long)(n0 + 32 + lane) * H + kpf, 0, 1);
    }
    v16bf b[4];
#pragma unroll
    for (int ni = 0; ni < 4; ++ni) b[ni] = load_b_bf16(Wob, H, n0 + 16 * ni, k0);
#pragma unroll
    for (int mi = 0; mi < 4; ++mi) {
      v16bf a = load_a_bf16(outb, H, m0 + 16 * mi, k0);
#pragma unroll
      for (int ni = 0; ni < 4; ++ni) c[mi][ni] = wmma_bf16(a, b[ni], c[mi][ni]);
    }
  }
#pragma unroll
  for (int mi = 0; mi < 4; ++mi)
#pragma unroll
    for (int ni = 0; ni < 4; ++ni)
      store_c_f32_bias(y, H, m0 + 16 * mi, n0 + 16 * ni, c[mi][ni], bo);
}

// ---------------------------------------------------------------------------
extern "C" void kernel_launch(void* const* d_in, const int* in_sizes, int n_in,
                              void* d_out, int out_size, void* d_ws, size_t ws_size,
                              hipStream_t stream) {
  (void)in_sizes; (void)n_in; (void)out_size; (void)ws_size;
  const float* x  = (const float*)d_in[0];
  const float* Ws = (const float*)d_in[1];
  const float* Wa = (const float*)d_in[2];
  const float* Wo = (const float*)d_in[3];
  const float* bo = (const float*)d_in[4];
  float* y = (float*)d_out;

  constexpr int Bn = 4, L = 4096, H = 2048, R = 256;
  const float inv_scale = 1.0f / 16.0f;  // 1/sqrt(256)

  char* ws = (char*)d_ws;
  size_t off = 0;
  auto take = [&](size_t bytes) -> void* {
    void* p = ws + off;
    off += (bytes + 255) & ~(size_t)255;
    return p;
  };
  bf16_t* WsT     = (bf16_t*)take((size_t)R * H * 2);
  bf16_t* WaT     = (bf16_t*)take((size_t)R * H * 2);
  bf16_t* Wob     = (bf16_t*)take((size_t)H * H * 2);
  bf16_t* stateb  = (bf16_t*)take((size_t)Bn * L * R * 2);
  bf16_t* actionb = (bf16_t*)take((size_t)Bn * L * R * 2);
  bf16_t* xT      = (bf16_t*)take((size_t)H * L * 2);   // per-batch, reused
  bf16_t* Pb      = (bf16_t*)take((size_t)L * L * 2);   // per-batch, reused
  bf16_t* outb    = (bf16_t*)take((size_t)L * H * 2);   // per-batch, reused

  // Weight prep
  k_prep_wt<<<dim3((H * R + 255) / 256, 2), 256, 0, stream>>>(Ws, Wa, WsT, WaT, H, R);
  k_prep_wo<<<dim3((H * H + 255) / 256), 256, 0, stream>>>(Wo, Wob, H * H);

  // Projections over all batches at once (M = B*L)
  k_proj<<<dim3(Bn * L / 64, R / 64, 2), 32, 0, stream>>>(
      x, WsT, WaT, stateb, actionb, H, R, inv_scale);

  for (int b = 0; b < Bn; ++b) {
    const float* xb = x + (size_t)b * L * H;
    k_transpose<<<dim3(L / 32, H / 32), dim3(32, 8), 0, stream>>>(xb, xT, H, L);
    k_scores<<<dim3(L / 16), 128, 0, stream>>>(
        stateb + (size_t)b * L * R, actionb + (size_t)b * L * R, Pb, L, R);
    k_pv<<<dim3(L / 64, H / 64), 32, 0, stream>>>(Pb, xT, outb, L, H);
    k_out<<<dim3(L / 64, H / 64), 32, 0, stream>>>(
        outb, Wob, bo, y + (size_t)b * L * H, H);
  }
}
